// NodeBlockGVP_83786222011222
// MI455X (gfx1250) — matrix operand order repaired and checked by reading, not compiled, not run on hardware
//
#include <hip/hip_runtime.h>
#include <hip/hip_bf16.h>
#include <stdint.h>

typedef __bf16 bf16;
typedef bf16  v16bf __attribute__((ext_vector_type(16)));
typedef float v8f   __attribute__((ext_vector_type(8)));
typedef int   v4i   __attribute__((ext_vector_type(4)));

union Frag { v16bf bf; v4i q[2]; };

struct GvpWeights {
  const float* wh;     // VI x H
  const float* ws_b;   // 128
  const float* wv;     // H x VO
  const float* wsv_w;  // 128 x VO
  const float* wsv_b;  // VO
  const bf16*  wsT;    // [128][KPAD] bf16, transposed ws_w, zero padded
};

#define KPAD 192
#define SOUT 128
#define EPSV 1e-5f

__device__ __forceinline__ float sigm(float x) { return 1.f / (1.f + __expf(-x)); }

// One wave (32 lanes) processes a tile of 16 items.
// s_in: 16 x SI (row stride SI), v_in: 16 x VI x 3 (row stride VI*3)
// s_out: 16 x 128, v_out: 16 x VO x 3
// vhb: >= 16*H*3 floats scratch, vgb: >= 16*H floats scratch, act: 16*KPAD bf16
template<int SI, int VI, int VO, bool RELU>
__device__ __forceinline__ void gvp_tile(
    int lane, const GvpWeights& W,
    const float* s_in, const float* v_in,
    float* s_out, float* v_out,
    float* vhb, float* vgb, bf16* act)
{
  constexpr int H  = (VI > VO) ? VI : VO;
  constexpr int K  = SI + H;
  constexpr int KT = (K + 31) / 32;

  // 1) vh[i][h][c] = sum_v v_in[i][v][c] * wh[v][h]
  for (int idx = lane; idx < 16 * H * 3; idx += 32) {
    int i = idx / (H * 3);
    int r = idx - i * (H * 3);
    int h = r / 3, c = r - h * 3;
    float a = 0.f;
#pragma unroll
    for (int v = 0; v < VI; ++v)
      a += v_in[(i * VI + v) * 3 + c] * W.wh[v * H + h];
    vhb[(i * H + h) * 3 + c] = a;
  }
  __syncthreads();

  // 2) vn = sqrt(clamp(|vh|^2, eps))
  for (int idx = lane; idx < 16 * H; idx += 32) {
    float x = vhb[idx * 3 + 0], y = vhb[idx * 3 + 1], z = vhb[idx * 3 + 2];
    vgb[idx] = sqrtf(fmaxf(x * x + y * y + z * z, EPSV));
  }
  __syncthreads();

  // 3) bf16 activation tile [16][KPAD] = concat(s, vn), zero padded
  for (int idx = lane; idx < 16 * KPAD; idx += 32) {
    int i = idx / KPAD, k = idx - i * KPAD;
    float v = 0.f;
    if (k < SI)          v = s_in[i * SI + k];
    else if (k < SI + H) v = vgb[i * H + (k - SI)];
    act[idx] = (bf16)v;
  }
  __syncthreads();

  // 4) s_lin(16x128) = act(16xK) @ ws(Kx128) via v_wmma_f32_16x16x32_bf16
  v8f acc[8];
  {
    v8f z = {0.f, 0.f, 0.f, 0.f, 0.f, 0.f, 0.f, 0.f};
#pragma unroll
    for (int nt = 0; nt < 8; ++nt) acc[nt] = z;
  }
  const int arow  = lane & 15;
  const int ahalf = lane >> 4;
#pragma unroll
  for (int kt = 0; kt < KT; ++kt) {
    // A frag: lanes 0-15 -> row=lane, K 0-7 & 16-23; lanes 16-31 -> K 8-15 & 24-31
    Frag A;
    const v4i* ap = (const v4i*)(act + arow * KPAD + kt * 32 + ahalf * 8);
    A.q[0] = ap[0];
    A.q[1] = ap[2];   // +16 elements = +32 bytes
#pragma unroll
    for (int nt = 0; nt < 8; ++nt) {
      // B frag: lanes 0-15 -> col, K 0-7 & 16-23; lanes 16-31 -> col, K 8-15 & 24-31
      Frag B;
      const v4i* bp = (const v4i*)(W.wsT + (size_t)(nt * 16 + arow) * KPAD + kt * 32 + ahalf * 8);
      B.q[0] = bp[0];
      B.q[1] = bp[2];
      acc[nt] = __builtin_amdgcn_wmma_f32_16x16x32_bf16(
          false, A.bf, false, B.bf, (short)0, acc[nt], false, false);
    }
  }

  // 5) s_out (pre-activation) = s_lin + bias  (C layout: VGPR r -> M=r(+8), N=lane(&15))
#pragma unroll
  for (int nt = 0; nt < 8; ++nt) {
    int n = nt * 16 + arow;
    float b = W.ws_b[n];
#pragma unroll
    for (int r = 0; r < 8; ++r) {
      int m = r + ahalf * 8;
      s_out[m * SOUT + n] = acc[nt][r] + b;
    }
  }
  __syncthreads();

  // 6) gate[i][o] = sigmoid( sigmoid(s_out) @ wsv_w + wsv_b )
  for (int idx = lane; idx < 16 * VO; idx += 32) {
    int i = idx / VO, o = idx - i * VO;
    float g = W.wsv_b[o];
    for (int so = 0; so < SOUT; ++so)
      g += sigm(s_out[i * SOUT + so]) * W.wsv_w[so * VO + o];
    vgb[idx] = sigm(g);
  }
  __syncthreads();

  // 7) v_out = (vh @ wv) * gate
  for (int idx = lane; idx < 16 * VO * 3; idx += 32) {
    int i = idx / (VO * 3);
    int r = idx - i * (VO * 3);
    int o = r / 3, c = r - o * 3;
    float a = 0.f;
#pragma unroll
    for (int h = 0; h < H; ++h)
      a += vhb[(i * H + h) * 3 + c] * W.wv[h * VO + o];
    v_out[(i * VO + o) * 3 + c] = a * vgb[i * VO + o];
  }

  // 8) optional ReLU on scalar output
  if (RELU) {
    for (int idx = lane; idx < 16 * SOUT; idx += 32)
      s_out[idx] = fmaxf(s_out[idx], 0.f);
  }
  __syncthreads();
}

#define DECL_TILE_SMEM() \
  __shared__ __attribute__((aligned(16))) float sA[16 * 144];   \
  __shared__ __attribute__((aligned(16))) float sB[16 * 144];   \
  __shared__ __attribute__((aligned(16))) float vA[16 * 96];    \
  __shared__ __attribute__((aligned(16))) float vB[16 * 96];    \
  __shared__ __attribute__((aligned(16))) float vhs[16 * 96];   \
  __shared__ __attribute__((aligned(16))) float vgs[16 * 32];   \
  __shared__ __attribute__((aligned(16))) bf16  acts[16 * KPAD];

// ---------------- node kernel: centroid + node0 + node1 ----------------
__global__ __launch_bounds__(32) void node_kernel(
    const float* __restrict__ x_s, const float* __restrict__ x_v,
    GvpWeights gc, GvpWeights g0, GvpWeights g1,
    float* __restrict__ cs, float* __restrict__ cv,
    float* __restrict__ hns, float* __restrict__ hnv, int N)
{
  DECL_TILE_SMEM();
  const int lane = threadIdx.x;
  const int n0 = blockIdx.x * 16;

  for (int idx = lane; idx < 16 * 128; idx += 32) {
    int i = idx >> 7, k = idx & 127;
    int n = min(n0 + i, N - 1);
    sA[i * 128 + k] = x_s[(size_t)n * 128 + k];
  }
  for (int idx = lane; idx < 16 * 48; idx += 32) {
    int i = idx / 48, r = idx - i * 48;
    int n = min(n0 + i, N - 1);
    vA[i * 48 + r] = x_v[(size_t)n * 48 + r];
  }
  __syncthreads();

  // centroid: A -> B, store to global
  gvp_tile<128, 16, 16, false>(lane, gc, sA, vA, sB, vB, vhs, vgs, acts);
  for (int idx = lane; idx < 16 * 128; idx += 32) {
    int i = idx >> 7, k = idx & 127; int n = n0 + i;
    if (n < N) cs[(size_t)n * 128 + k] = sB[i * 128 + k];
  }
  for (int idx = lane; idx < 16 * 48; idx += 32) {
    int i = idx / 48, r = idx - i * 48; int n = n0 + i;
    if (n < N) cv[(size_t)n * 48 + r] = vB[i * 48 + r];
  }
  __syncthreads();

  // node0 (relu): A -> B ; node1: B -> A
  gvp_tile<128, 16, 16, true >(lane, g0, sA, vA, sB, vB, vhs, vgs, acts);
  gvp_tile<128, 16, 16, false>(lane, g1, sB, vB, sA, vA, vhs, vgs, acts);

  for (int idx = lane; idx < 16 * 128; idx += 32) {
    int i = idx >> 7, k = idx & 127; int n = n0 + i;
    if (n < N) hns[(size_t)n * 128 + k] = sA[i * 128 + k];
  }
  for (int idx = lane; idx < 16 * 48; idx += 32) {
    int i = idx / 48, r = idx - i * 48; int n = n0 + i;
    if (n < N) hnv[(size_t)n * 48 + r] = vA[i * 48 + r];
  }
}

// ---------------- edge kernel: edge0+edge1 -> gather/fuse -> msg -> scatter ----------------
__global__ __launch_bounds__(32) void edge_kernel(
    const float* __restrict__ eas, const float* __restrict__ eav,
    const int* __restrict__ eidx,
    GvpWeights ge0, GvpWeights ge1, GvpWeights gm,
    const float* __restrict__ hns, const float* __restrict__ hnv,
    float* __restrict__ accs, float* __restrict__ accv, int E)
{
  DECL_TILE_SMEM();
  const int lane = threadIdx.x;
  const int e0 = blockIdx.x * 16;

  for (int idx = lane; idx < 16 * 64; idx += 32) {
    int i = idx >> 6, k = idx & 63;
    int e = min(e0 + i, E - 1);
    sA[i * 64 + k] = eas[(size_t)e * 64 + k];
  }
  for (int idx = lane; idx < 16 * 48; idx += 32) {
    int i = idx / 48, r = idx - i * 48;
    int e = min(e0 + i, E - 1);
    vA[i * 48 + r] = eav[(size_t)e * 48 + r];
  }
  __syncthreads();

  gvp_tile<64, 16, 16, true >(lane, ge0, sA, vA, sB, vB, vhs, vgs, acts);
  gvp_tile<128, 16, 16, false>(lane, ge1, sB, vB, sA, vA, vhs, vgs, acts);

  // fuse(h_edge, h_node[col]) -> sB (16x144), vB (16x32x3)
  for (int idx = lane; idx < 16 * 128; idx += 32) {
    int i = idx >> 7, k = idx & 127;
    int e = min(e0 + i, E - 1); int c = eidx[E + e];
    sB[i * 144 + k] = sA[i * 128 + k] * hns[(size_t)c * 128 + k];
  }
  for (int idx = lane; idx < 256; idx += 32) {       // dot(lv, rv) over xyz
    int i = idx >> 4, o = idx & 15;
    int e = min(e0 + i, E - 1); int c = eidx[E + e];
    float d = 0.f;
#pragma unroll
    for (int cc = 0; cc < 3; ++cc)
      d += vA[i * 48 + o * 3 + cc] * hnv[(size_t)c * 48 + o * 3 + cc];
    sB[i * 144 + 128 + o] = d;
  }
  for (int idx = lane; idx < 768; idx += 32) {       // (lv + rv) * 0.5
    int i = idx / 48, r = idx - i * 48;
    int e = min(e0 + i, E - 1); int c = eidx[E + e];
    vB[i * 96 + r] = 0.5f * (vA[i * 48 + r] + hnv[(size_t)c * 48 + r]);
  }
  for (int idx = lane; idx < 256; idx += 32) {       // cross(lv, rv)
    int i = idx >> 4, o = idx & 15;
    int e = min(e0 + i, E - 1); int c = eidx[E + e];
    float lx = vA[i * 48 + o * 3 + 0], ly = vA[i * 48 + o * 3 + 1], lz = vA[i * 48 + o * 3 + 2];
    float rx = hnv[(size_t)c * 48 + o * 3 + 0];
    float ry = hnv[(size_t)c * 48 + o * 3 + 1];
    float rz = hnv[(size_t)c * 48 + o * 3 + 2];
    vB[i * 96 + 48 + o * 3 + 0] = ly * rz - lz * ry;
    vB[i * 96 + 48 + o * 3 + 1] = lz * rx - lx * rz;
    vB[i * 96 + 48 + o * 3 + 2] = lx * ry - ly * rx;
  }
  __syncthreads();

  gvp_tile<144, 32, 16, false>(lane, gm, sB, vB, sA, vA, vhs, vgs, acts);

  // scatter-sum to source nodes (row = eidx[e]); native fp32 global atomics
  for (int idx = lane; idx < 16 * 128; idx += 32) {
    int i = idx >> 7, k = idx & 127;
    int e = e0 + i;
    if (e < E) {
      int r = eidx[e];
      unsafeAtomicAdd(accs + (size_t)r * 128 + k, sA[i * 128 + k]);
    }
  }
  for (int idx = lane; idx < 16 * 48; idx += 32) {
    int i = idx / 48, r = idx - i * 48;
    int e = e0 + i;
    if (e < E) {
      int rr = eidx[e];
      unsafeAtomicAdd(accv + (size_t)rr * 48 + r, vA[i * 48 + r]);
    }
  }
}

// ---------------- out kernel: fuse -> out0 -> out1 -> residual + LayerNorms ----------------
__global__ __launch_bounds__(32) void out_kernel(
    const float* __restrict__ x_s, const float* __restrict__ x_v,
    const float* __restrict__ cs, const float* __restrict__ cv,
    const float* __restrict__ accs, const float* __restrict__ accv,
    GvpWeights go0, GvpWeights go1,
    const float* __restrict__ ln_w, const float* __restrict__ ln_b,
    float* __restrict__ outS, float* __restrict__ outV, int N)
{
  DECL_TILE_SMEM();
  const int lane = threadIdx.x;
  const int n0 = blockIdx.x * 16;

  for (int idx = lane; idx < 16 * 128; idx += 32) {
    int i = idx >> 7, k = idx & 127;
    int n = min(n0 + i, N - 1);
    sA[i * 144 + k] = cs[(size_t)n * 128 + k];
    sB[i * 128 + k] = accs[(size_t)n * 128 + k];
  }
  for (int idx = lane; idx < 768; idx += 32) {
    int i = idx / 48, r = idx - i * 48;
    int n = min(n0 + i, N - 1);
    vA[i * 96 + r] = cv[(size_t)n * 48 + r];
    vB[i * 48 + r] = accv[(size_t)n * 48 + r];
  }
  __syncthreads();

  // fuse(cs,cv, as,av) in place: first dot & cross (read-only on vA[0..15])
  for (int idx = lane; idx < 256; idx += 32) {
    int i = idx >> 4, o = idx & 15;
    float d = 0.f;
#pragma unroll
    for (int cc = 0; cc < 3; ++cc)
      d += vA[i * 96 + o * 3 + cc] * vB[i * 48 + o * 3 + cc];
    sA[i * 144 + 128 + o] = d;
  }
  for (int idx = lane; idx < 256; idx += 32) {
    int i = idx >> 4, o = idx & 15;
    float lx = vA[i * 96 + o * 3 + 0], ly = vA[i * 96 + o * 3 + 1], lz = vA[i * 96 + o * 3 + 2];
    float rx = vB[i * 48 + o * 3 + 0], ry = vB[i * 48 + o * 3 + 1], rz = vB[i * 48 + o * 3 + 2];
    vA[i * 96 + 48 + o * 3 + 0] = ly * rz - lz * ry;
    vA[i * 96 + 48 + o * 3 + 1] = lz * rx - lx * rz;
    vA[i * 96 + 48 + o * 3 + 2] = lx * ry - ly * rx;
  }
  __syncthreads();
  for (int idx = lane; idx < 768; idx += 32) {
    int i = idx / 48, r = idx - i * 48;
    vA[i * 96 + r] = 0.5f * (vA[i * 96 + r] + vB[i * 48 + r]);
  }
  for (int idx = lane; idx < 16 * 128; idx += 32) {
    int i = idx >> 7, k = idx & 127;
    sA[i * 144 + k] *= sB[i * 128 + k];
  }
  __syncthreads();

  gvp_tile<144, 32, 16, true >(lane, go0, sA, vA, sB, vB, vhs, vgs, acts);
  gvp_tile<128, 16, 16, false>(lane, go1, sB, vB, sA, vA, vhs, vgs, acts);

  // residuals
  for (int idx = lane; idx < 16 * 128; idx += 32) {
    int i = idx >> 7, k = idx & 127;
    int n = min(n0 + i, N - 1);
    sA[i * 128 + k] += x_s[(size_t)n * 128 + k];
  }
  for (int idx = lane; idx < 768; idx += 32) {
    int i = idx / 48, r = idx - i * 48;
    int n = min(n0 + i, N - 1);
    vA[i * 48 + r] += x_v[(size_t)n * 48 + r];
  }
  __syncthreads();

  // scalar LN stats (one lane per item)
  if (lane < 16) {
    float s = 0.f, s2 = 0.f;
    for (int k = 0; k < 128; ++k) {
      float v = sA[lane * 128 + k];
      s += v; s2 += v * v;
    }
    float mu = s * (1.f / 128.f);
    vgs[lane] = mu;
    vgs[16 + lane] = s2 * (1.f / 128.f) - mu * mu;
  }
  // vector LN stats per (item, xyz)
  for (int idx = lane; idx < 48; idx += 32) {
    int i = idx / 3, c = idx - i * 3;
    float s = 0.f, s2 = 0.f;
    for (int o = 0; o < 16; ++o) {
      float v = vA[i * 48 + o * 3 + c];
      s += v; s2 += v * v;
    }
    float mu = s * (1.f / 16.f);
    vhs[idx] = mu;
    vhs[48 + idx] = s2 * (1.f / 16.f) - mu * mu;
  }
  __syncthreads();

  for (int idx = lane; idx < 16 * 128; idx += 32) {
    int i = idx >> 7, k = idx & 127;
    int n = n0 + i;
    if (n < N)
      outS[(size_t)n * 128 + k] =
          (sA[i * 128 + k] - vgs[i]) * rsqrtf(vgs[16 + i] + EPSV) * ln_w[k] + ln_b[k];
  }
  for (int idx = lane; idx < 768; idx += 32) {
    int i = idx / 48, r = idx - i * 48;
    int c = r % 3;
    int n = n0 + i;
    if (n < N)
      outV[(size_t)n * 48 + r] =
          (vA[i * 48 + r] - vhs[i * 3 + c]) * rsqrtf(vhs[48 + i * 3 + c] + EPSV);
  }
}

// ---------------- helpers ----------------
struct PrepArgs { const float* w[8]; int K[8]; };

__global__ void prep_weights(PrepArgs pa, bf16* __restrict__ dst)
{
  int g = blockIdx.x;
  const float* w = pa.w[g];
  int K = pa.K[g];
  bf16* d = dst + (size_t)g * 128 * KPAD;
  for (int idx = threadIdx.x; idx < 128 * KPAD; idx += blockDim.x) {
    int o = idx / KPAD, k = idx - o * KPAD;
    float v = (k < K) ? w[(size_t)k * 128 + o] : 0.f;
    d[(size_t)o * KPAD + k] = (bf16)v;
  }
}

__global__ void zero_kernel(float* __restrict__ p, long n)
{
  long i = (long)blockIdx.x * blockDim.x + threadIdx.x;
  long stride = (long)gridDim.x * blockDim.x;
  for (; i < n; i += stride) p[i] = 0.f;
}

extern "C" void kernel_launch(void* const* d_in, const int* in_sizes, int n_in,
                              void* d_out, int out_size, void* d_ws, size_t ws_size,
                              hipStream_t stream)
{
  const float* x_s  = (const float*)d_in[0];
  const float* x_v  = (const float*)d_in[1];
  const float* eas  = (const float*)d_in[2];
  const float* eav  = (const float*)d_in[3];
  const int*   eidx = (const int*)d_in[4];
  const int N = in_sizes[0] / 128;
  const int E = in_sizes[2] / 64;

  auto P = [&](int g, int j) -> const float* { return (const float*)d_in[5 + g * 6 + j]; };
  const float* ln_w = (const float*)d_in[5 + 48];
  const float* ln_b = (const float*)d_in[5 + 49];

  char* base = (char*)d_ws;
  bf16* wsT = (bf16*)base;
  size_t off = (((size_t)8 * 128 * KPAD * sizeof(bf16)) + 255) & ~(size_t)255;
  float* cs   = (float*)(base + off); off += (size_t)N * 128 * 4;
  float* cv   = (float*)(base + off); off += (size_t)N * 48 * 4;
  float* hns  = (float*)(base + off); off += (size_t)N * 128 * 4;
  float* hnv  = (float*)(base + off); off += (size_t)N * 48 * 4;
  float* accs = (float*)(base + off); off += (size_t)N * 128 * 4;
  float* accv = (float*)(base + off); off += (size_t)N * 48 * 4;
  (void)ws_size; (void)n_in; (void)out_size;

  // param order: node0, node1, edge0, edge1, msg, centroid, out0, out1
  static const int Ks[8] = {144, 144, 80, 144, 176, 144, 176, 144};
  GvpWeights G[8];
  PrepArgs pa;
  for (int g = 0; g < 8; ++g) {
    G[g].wh    = P(g, 0);
    G[g].ws_b  = P(g, 2);
    G[g].wv    = P(g, 3);
    G[g].wsv_w = P(g, 4);
    G[g].wsv_b = P(g, 5);
    G[g].wsT   = wsT + (size_t)g * 128 * KPAD;
    pa.w[g] = P(g, 1);
    pa.K[g] = Ks[g];
  }

  prep_weights<<<8, 256, 0, stream>>>(pa, wsT);
  long zn = (long)N * 176;  // accs followed immediately by accv
  zero_kernel<<<2048, 256, 0, stream>>>(accs, zn);
  node_kernel<<<(N + 15) / 16, 32, 0, stream>>>(x_s, x_v, G[5], G[0], G[1],
                                                cs, cv, hns, hnv, N);
  edge_kernel<<<(E + 15) / 16, 32, 0, stream>>>(eas, eav, eidx, G[2], G[3], G[4],
                                                hns, hnv, accs, accv, E);
  out_kernel<<<(N + 15) / 16, 32, 0, stream>>>(x_s, x_v, cs, cv, accs, accv,
                                               G[6], G[7], ln_w, ln_b,
                                               (float*)d_out, (float*)d_out + (size_t)N * 128, N);
}
